// self_attention_37976100831498
// MI455X (gfx1250) — matrix operand order repaired and checked by reading, not compiled, not run on hardware
//
#include <hip/hip_runtime.h>

#define BATCH   8
#define SEQ     2048
#define DIM     128
#define BLOCK_M 64
#define KPAD    136   // 128 + 8 halfs: rotate LDS banks between rows
#define VPAD    40    // 32 + 8
#define PPAD    40    // 32 + 8

typedef __attribute__((ext_vector_type(16))) _Float16 v16h;
typedef __attribute__((ext_vector_type(8)))  _Float16 v8h;
typedef __attribute__((ext_vector_type(2)))  _Float16 v2h;
typedef __attribute__((ext_vector_type(8)))  float    v8f;

static __device__ __forceinline__ v16h cat8(v8h a, v8h b) {
  return __builtin_shufflevector(a, b, 0,1,2,3,4,5,6,7,8,9,10,11,12,13,14,15);
}

// load 8 contiguous f32, scale, convert to 8 f16
static __device__ __forceinline__ v8h ld8cvt(const float* p, float scale) {
  float4 a = ((const float4*)p)[0];
  float4 b = ((const float4*)p)[1];
  v8h r;
  r[0]=(_Float16)(a.x*scale); r[1]=(_Float16)(a.y*scale);
  r[2]=(_Float16)(a.z*scale); r[3]=(_Float16)(a.w*scale);
  r[4]=(_Float16)(b.x*scale); r[5]=(_Float16)(b.y*scale);
  r[6]=(_Float16)(b.z*scale); r[7]=(_Float16)(b.w*scale);
  return r;
}

// convert 8 f32 (already in registers) to 8 f16
static __device__ __forceinline__ v8h cvt8(const float* d) {
  v8h r;
  #pragma unroll
  for (int i = 0; i < 8; ++i) r[i] = (_Float16)d[i];
  return r;
}

static __device__ __forceinline__ void ld16(const float* p, float* d) {
  #pragma unroll
  for (int i = 0; i < 4; ++i) {
    float4 t = ((const float4*)p)[i];
    d[4*i+0]=t.x; d[4*i+1]=t.y; d[4*i+2]=t.z; d[4*i+3]=t.w;
  }
}

__global__ __launch_bounds__(128)
void fa_fwd_causal_wmma(const float* __restrict__ Q,
                        const float* __restrict__ K,
                        const float* __restrict__ V,
                        float* __restrict__ O) {
  // double-buffered K (row-major) and V^T tiles
  __shared__ _Float16 Klds[2 * 32 * KPAD];
  __shared__ _Float16 VTlds[2 * DIM * VPAD];
  __shared__ _Float16 Plds[4 * 16 * PPAD];   // per-wave P transpose buffer

  const int tile = blockIdx.x;              // query tile (0..31)
  const int b    = blockIdx.y;              // batch
  const int tid  = threadIdx.x;
  const int wave = tid >> 5;
  const int lane = tid & 31;
  const int g    = lane >> 4;               // lane group (WMMA K/M half)
  const int col  = lane & 15;               // WMMA N column index

  const int m0 = tile * BLOCK_M + wave * 16;
  // base-2 softmax: fold 1/sqrt(128) * log2(e) into Q so exp == bare v_exp_f32
  const float qscale = 0.08838834764831845f * 1.4426950408889634f;

  const size_t base = (size_t)b * SEQ * DIM;
  const float* Qp = Q + base;
  const float* Kp = K + base;
  const float* Vp = V + base;
  float*       Op = O + base;

  // ---- Q tile (16 rows x 128) -> A-layout f16 registers, pre-scaled ----
  v16h Qa[4];
  {
    const float* qrow = Qp + (size_t)(m0 + col) * DIM;
    #pragma unroll
    for (int c = 0; c < 4; ++c) {
      v8h lo = ld8cvt(qrow + 32*c + 16*g,     qscale);
      v8h hi = ld8cvt(qrow + 32*c + 16*g + 8, qscale);
      Qa[c] = cat8(lo, hi);
    }
  }

  // all-ones B tile: P x ones accumulates the softmax denominator l
  v16h onesB;
  #pragma unroll
  for (int i = 0; i < 16; ++i) onesB[i] = (_Float16)1.0f;

  // acc[0..7] = output d-tiles, acc[8] = l (row-sum) tile
  v8f acc[9];
  #pragma unroll
  for (int t = 0; t < 9; ++t)
    #pragma unroll
    for (int r = 0; r < 8; ++r) acc[t][r] = 0.f;

  float mrow[8];
  #pragma unroll
  for (int r = 0; r < 8; ++r) mrow[r] = -__builtin_inff();

  const int nkb = 2 * (tile + 1);    // 32-key blocks covering causal range
  const int pp  = tid >> 3;          // staging: key-row pair 0..15
  const int cg  = tid & 7;           // staging: 16-col group

  // ---- register prefetch buffers: 2 K rows + 2 V rows x 16 cols (f32) ----
  float kr0[16], kr1[16], vr0[16], vr1[16];
  {
    const float* k0 = Kp + (size_t)(2*pp) * DIM + 16*cg;
    const float* v0 = Vp + (size_t)(2*pp) * DIM + 16*cg;
    ld16(k0,       kr0); ld16(k0 + DIM, kr1);
    ld16(v0,       vr0); ld16(v0 + DIM, vr1);
  }

  for (int jb = 0; jb < nkb; ++jb) {
    const int keybase = jb * 32;
    const int buf     = jb & 1;
    _Float16* Kb_lds  = &Klds[buf * 32 * KPAD];
    _Float16* VT_lds  = &VTlds[buf * DIM * VPAD];

    // ---- convert prefetched block and store to LDS buffer `buf` ----
    {
      #pragma unroll
      for (int rr = 0; rr < 2; ++rr) {
        const float* src = rr ? kr1 : kr0;
        *(v8h*)&Kb_lds[(2*pp+rr)*KPAD + 16*cg]     = cvt8(src);
        *(v8h*)&Kb_lds[(2*pp+rr)*KPAD + 16*cg + 8] = cvt8(src + 8);
      }
      #pragma unroll
      for (int j = 0; j < 16; ++j) {
        v2h t;
        t[0] = (_Float16)vr0[j];
        t[1] = (_Float16)vr1[j];
        *(v2h*)&VT_lds[(16*cg + j)*VPAD + 2*pp] = t;
      }
    }

    __syncthreads();   // buf staged; also orders prior buffer's readers

    // ---- prefetch NEXT block from global (overlaps compute below) ----
    if (jb + 1 < nkb) {
      const int nb = (jb + 1) * 32;
      const float* k0 = Kp + (size_t)(nb + 2*pp) * DIM + 16*cg;
      const float* v0 = Vp + (size_t)(nb + 2*pp) * DIM + 16*cg;
      ld16(k0,       kr0); ld16(k0 + DIM, kr1);
      ld16(v0,       vr0); ld16(v0 + DIM, vr1);
    }

    if (keybase > m0 + 15) continue;   // wave-uniform: block fully masked

    // ---- scores: two 16x16 tiles, 4 chained WMMAs over D each ----
    v8f s0 = {0,0,0,0,0,0,0,0};
    v8f s1 = {0,0,0,0,0,0,0,0};
    {
      const _Float16* kb0 = &Kb_lds[(col)      * KPAD + 16*g];
      const _Float16* kb1 = &Kb_lds[(16 + col) * KPAD + 16*g];
      #pragma unroll
      for (int c = 0; c < 4; ++c) {
        v16h Kb = cat8(*(const v8h*)(kb0 + 32*c), *(const v8h*)(kb0 + 32*c + 8));
        s0 = __builtin_amdgcn_wmma_f32_16x16x32_f16(false, Qa[c], false, Kb,
                                                    (short)0, s0, false, false);
      }
      #pragma unroll
      for (int c = 0; c < 4; ++c) {
        v16h Kb = cat8(*(const v8h*)(kb1 + 32*c), *(const v8h*)(kb1 + 32*c + 8));
        s1 = __builtin_amdgcn_wmma_f32_16x16x32_f16(false, Qa[c], false, Kb,
                                                    (short)0, s1, false, false);
      }
    }

    // ---- causal mask: only diagonal blocks need it (wave-uniform test) ----
    const int rbase = m0 + 8*g;
    if (keybase + 31 > m0) {
      #pragma unroll
      for (int r = 0; r < 8; ++r) {
        const int rg = rbase + r;
        s0[r] = (keybase + col      <= rg) ? s0[r] : -__builtin_inff();
        s1[r] = (keybase + 16 + col <= rg) ? s1[r] : -__builtin_inff();
      }
    }

    // ---- online softmax (base 2): row max via half-wave shuffles ----
    float mnew[8];
    #pragma unroll
    for (int r = 0; r < 8; ++r)
      mnew[r] = fmaxf(fmaxf(s0[r], s1[r]), mrow[r]);
    #pragma unroll
    for (int msk = 1; msk < 16; msk <<= 1)      // stays inside 16-lane half
      #pragma unroll
      for (int r = 0; r < 8; ++r)
        mnew[r] = fmaxf(mnew[r], __shfl_xor(mnew[r], msk, 32));

    v8f alphav;
    #pragma unroll
    for (int r = 0; r < 8; ++r) {
      alphav[r] = __builtin_amdgcn_exp2f(mrow[r] - mnew[r]);
      mrow[r]   = mnew[r];
      s0[r]     = __builtin_amdgcn_exp2f(s0[r] - mnew[r]);
      s1[r]     = __builtin_amdgcn_exp2f(s1[r] - mnew[r]);
    }
    #pragma unroll
    for (int t = 0; t < 9; ++t)        // vector mul -> v_pk_mul_f32
      acc[t] = acc[t] * alphav;

    // ---- P: C layout -> LDS -> A layout; P·V (8) + P·1 (l tile) ----
    {
      _Float16* pw = &Plds[wave * 16 * PPAD];
      #pragma unroll
      for (int r = 0; r < 8; ++r) {
        const int row = r + 8*g;
        pw[row * PPAD + col]      = (_Float16)s0[r];
        pw[row * PPAD + 16 + col] = (_Float16)s1[r];
      }
      // same-wave LDS is in-order: no barrier needed for this private buffer
      const _Float16* pr = &Plds[wave * 16 * PPAD + col * PPAD + 16*g];
      v16h Pa = cat8(*(const v8h*)pr, *(const v8h*)(pr + 8));

      #pragma unroll
      for (int dt = 0; dt < 8; ++dt) {
        const _Float16* vb = &VT_lds[(16*dt + col) * VPAD + 16*g];
        v16h Vb = cat8(*(const v8h*)vb, *(const v8h*)(vb + 8));
        acc[dt] = __builtin_amdgcn_wmma_f32_16x16x32_f16(false, Pa, false, Vb,
                                                         (short)0, acc[dt],
                                                         false, false);
      }
      acc[8] = __builtin_amdgcn_wmma_f32_16x16x32_f16(false, Pa, false, onesB,
                                                      (short)0, acc[8],
                                                      false, false);
    }
  }

  // ---- epilogue: normalize rows by l (acc[8]) and store coalesced ----
  #pragma unroll
  for (int r = 0; r < 8; ++r) {
    const float inv = 1.0f / acc[8][r];
    float* orow = Op + (size_t)(m0 + 8*g + r) * DIM + col;
    #pragma unroll
    for (int dt = 0; dt < 8; ++dt)
      orow[16*dt] = acc[dt][r] * inv;
  }
}

extern "C" void kernel_launch(void* const* d_in, const int* in_sizes, int n_in,
                              void* d_out, int out_size, void* d_ws, size_t ws_size,
                              hipStream_t stream) {
  (void)in_sizes; (void)n_in; (void)out_size; (void)d_ws; (void)ws_size;
  const float* Q = (const float*)d_in[0];
  const float* K = (const float*)d_in[1];
  const float* V = (const float*)d_in[2];
  float* O = (float*)d_out;
  dim3 grid(SEQ / BLOCK_M, BATCH);
  fa_fwd_causal_wmma<<<grid, 128, 0, stream>>>(Q, K, V, O);
}
